// TransformerBlock_54339926229272
// MI455X (gfx1250) — matrix operand order repaired and checked by reading, not compiled, not run on hardware
//
#include <hip/hip_runtime.h>
#include <hip/hip_bf16.h>

// ---------------------------------------------------------------------------
// Swin transformer block on gfx1250 (MI455X), bf16 WMMA pipeline.
//   - all GEMMs: v_wmma_f32_16x16x32_bf16, 32(M) x 64(N) tile per wave
//   - attention: per (window, head, 16-row tile) wave; V stored transposed so
//     both K and V B-fragments are single contiguous 32B loads per lane
//   - precomputed index maps remove all integer division from epilogues
// ---------------------------------------------------------------------------

typedef __attribute__((ext_vector_type(16))) __bf16 bf16x16;
typedef __attribute__((ext_vector_type(8)))  __bf16 bf16x8;
typedef __attribute__((ext_vector_type(8)))  float  f32x8;

#define DEV static __device__ __forceinline__

// Model constants
#define Bz_     32
#define Hh_     56
#define DIM_    256
#define HEADS_  8
#define HD_     32
#define WS_     7
#define N_      49
#define NW_     64
#define SHIFT_  3
#define HID_    1024
#define TOK_    100352      // Bz * 56 * 56
#define WT_     2048        // Bz * NW

DEV unsigned short f2bf(float f) {
    unsigned int u = __float_as_uint(f);
    u += 0x7FFFu + ((u >> 16) & 1u);          // round-to-nearest-even
    return (unsigned short)(u >> 16);
}

DEV float wave_red_add32(float v) {
    #pragma unroll
    for (int m = 16; m >= 1; m >>= 1) v += __shfl_xor(v, m, 32);
    return v;
}
DEV float half_red_add(float v) {             // reduce within 16-lane half
    #pragma unroll
    for (int m = 8; m >= 1; m >>= 1) v += __shfl_xor(v, m, 32);
    return v;
}
DEV float half_red_max(float v) {
    #pragma unroll
    for (int m = 8; m >= 1; m >>= 1) v = fmaxf(v, __shfl_xor(v, m, 32));
    return v;
}

// ---------------------------------------------------------------------------
// Core: one wave computes a (MT*16) x 64 f32 tile of  A(MxK,bf16) @ W(NxK)^T
// A row-major (lda = K), W row-major (N x K)  -> B[k][n] = W[n][k].
// Fragment layouts as used here (A and B share contiguous k order):
//   A 16x32 bf16: lane l -> row l&15; half j = K[kbA + j], kbA = (l>>4)*16
//   B 32x16 bf16: lane l -> col l&15; half j = K[kbB + j], kbB = (l>>4)*16
//   C/D f32:      vgpr r, lane l -> row r + 8*(l>>4), col l&15
// ---------------------------------------------------------------------------
template <int K, int MT>
DEV void gemm_tile(const unsigned short* __restrict__ A,
                   const unsigned short* __restrict__ W,
                   f32x8 acc[MT][4], int lane) {
    const int row = lane & 15;
    const int hi  = lane >> 4;
    for (int k0 = 0; k0 < K; k0 += 32) {
        bf16x16 a[MT];
        #pragma unroll
        for (int mi = 0; mi < MT; mi++) {
            bf16x8 alo = *reinterpret_cast<const bf16x8*>(A + (mi * 16 + row) * K + k0 + hi * 16);
            bf16x8 ahi = *reinterpret_cast<const bf16x8*>(A + (mi * 16 + row) * K + k0 + 8 + hi * 16);
            #pragma unroll
            for (int i = 0; i < 8; i++) { a[mi][i] = alo[i]; a[mi][i + 8] = ahi[i]; }
        }
        #pragma unroll
        for (int t = 0; t < 4; t++) {
            bf16x16 b = *reinterpret_cast<const bf16x16*>(W + (t * 16 + row) * K + k0 + hi * 16);
            #pragma unroll
            for (int mi = 0; mi < MT; mi++)
                acc[mi][t] = __builtin_amdgcn_wmma_f32_16x16x32_bf16(
                    false, a[mi], false, b, (short)0, acc[mi][t], false, false);
        }
    }
}

// ---------------------------------------------------------------------------
// K-1: precompute index maps (runs once per call, 392 blocks)
//   wn[m]   = (window << 6) | token-in-window          (windowed row decode)
//   omap[m] = original token index after +SHIFT roll    (gather/scatter map)
// ---------------------------------------------------------------------------
__global__ void k_maps(int* __restrict__ wn, int* __restrict__ omap) {
    int m = blockIdx.x * 256 + threadIdx.x;
    if (m >= TOK_) return;
    int w  = m / N_,  n  = m % N_;
    int bz = w >> 6,  wl = w & 63;
    int wh = wl >> 3, ww = wl & 7;
    int r  = n / WS_, c  = n % WS_;
    int sh = (wh * WS_ + r + SHIFT_) % Hh_;
    int sw = (ww * WS_ + c + SHIFT_) % Hh_;
    wn[m]   = (w << 6) | n;
    omap[m] = bz * (Hh_ * Hh_) + sh * Hh_ + sw;
}

// ---------------------------------------------------------------------------
// K0: f32 -> bf16 weight conversion
// ---------------------------------------------------------------------------
__global__ void k_cvt(const float* __restrict__ s, unsigned short* __restrict__ d, int n) {
    int i = blockIdx.x * 256 + threadIdx.x;
    if (i < n) d[i] = f2bf(s[i]);
}

// ---------------------------------------------------------------------------
// K1/K5: LayerNorm over DIM=256, one wave per token (8 f32 per lane),
// optional shift+window-partition gather via omap; bf16 output.
// ---------------------------------------------------------------------------
__global__ void k_ln(const float* __restrict__ x, const float* __restrict__ g,
                     const float* __restrict__ b, unsigned short* __restrict__ out,
                     const int* __restrict__ omap, int shifted) {
    int wave = blockIdx.x * (blockDim.x >> 5) + (threadIdx.x >> 5);
    int lane = threadIdx.x & 31;
    if (wave >= TOK_) return;
    size_t src_t = shifted ? (size_t)omap[wave] : (size_t)wave;
    const float* src = x + src_t * DIM_;
    float v[8];
    #pragma unroll
    for (int i = 0; i < 8; i++) v[i] = src[lane * 8 + i];
    float s = 0.f;
    #pragma unroll
    for (int i = 0; i < 8; i++) s += v[i];
    float mu = wave_red_add32(s) * (1.0f / DIM_);
    float vv = 0.f;
    #pragma unroll
    for (int i = 0; i < 8; i++) { float d = v[i] - mu; vv += d * d; }
    float inv = rsqrtf(wave_red_add32(vv) * (1.0f / DIM_) + 1e-5f);
    unsigned short* dst = out + (size_t)wave * DIM_;
    #pragma unroll
    for (int i = 0; i < 8; i++) {
        int ch = lane * 8 + i;
        dst[ch] = f2bf((v[i] - mu) * inv * g[ch] + b[ch]);
    }
}

// ---------------------------------------------------------------------------
// K2: QKV GEMM (100352x256 @ 256x768), 32x64 tile/wave; scatter epilogue into
// padded (win, head, 64, 32) Q/K and transposed (win, head, 32, 64) V.
// which-select (q/k/v) is wave-uniform: the 64-wide N tile never straddles a
// 256-column boundary. Row decode via precomputed wn[] (no division).
// Layout flats: QK dest = w*16384 + head*2048 + n*32 + d
//               Vt dest = w*16384 + head*2048 + d*64  + n
// ---------------------------------------------------------------------------
__global__ void k_qkv(const unsigned short* __restrict__ xw,
                      const unsigned short* __restrict__ Wq,
                      const float* __restrict__ bias,
                      const int* __restrict__ wn,
                      unsigned short* __restrict__ Qb,
                      unsigned short* __restrict__ Kb,
                      unsigned short* __restrict__ Vt) {
    int wave = blockIdx.x * 8 + (threadIdx.x >> 5);
    int lane = threadIdx.x & 31;
    int nt = wave % 12, mt = wave / 12;           // 12 = 768/64, mt < 3136
    int m0 = mt * 32, n0 = nt * 64;
    f32x8 acc[2][4] = {};
    gemm_tile<DIM_, 2>(xw + (size_t)m0 * DIM_, Wq + (size_t)n0 * DIM_, acc, lane);
    int row = lane & 15, hi = lane >> 4;
    int which = n0 >> 8;                          // wave-uniform: 0=Q 1=K 2=V
    int jj0 = (n0 & 255) + row;                   // col within 256 at t=0
    #pragma unroll
    for (int mi = 0; mi < 2; mi++)
        #pragma unroll
        for (int r = 0; r < 8; r++) {
            int m   = m0 + mi * 16 + r + 8 * hi;  // windowed token row
            int wnm = wn[m];
            int w = wnm >> 6, n = wnm & 63;
            if (which == 0) {
                int base = w * 16384 + n * HD_;
                #pragma unroll
                for (int t = 0; t < 4; t++) {
                    int jj = jj0 + t * 16;        // head*32 + d
                    float v = acc[mi][t][r] + bias[n0 + t * 16 + row];
                    Qb[base + (jj >> 5) * 2048 + (jj & 31)] = f2bf(v * 0.17677669529663689f);
                }
            } else if (which == 1) {
                int base = w * 16384 + n * HD_;
                #pragma unroll
                for (int t = 0; t < 4; t++) {
                    int jj = jj0 + t * 16;
                    float v = acc[mi][t][r] + bias[n0 + t * 16 + row];
                    Kb[base + (jj >> 5) * 2048 + (jj & 31)] = f2bf(v);
                }
            } else {
                int base = w * 16384 + n;
                #pragma unroll
                for (int t = 0; t < 4; t++) {
                    int jj = jj0 + t * 16;
                    float v = acc[mi][t][r] + bias[n0 + t * 16 + row];
                    Vt[base + (jj >> 5) * 2048 + (jj & 31) * 64] = f2bf(v);   // transposed
                }
            }
        }
}

// ---------------------------------------------------------------------------
// K3: windowed attention. One wave per (window, head, 16-row M-tile).
// S = Q K^T (4 WMMA, K=32=head_dim) + bias + mask, softmax via half-wave
// shuffles, P staged through LDS (C-layout -> A-fragment reshape),
// O = P V (4 WMMA, K=64 padded keys) with V read from transposed layout.
// ---------------------------------------------------------------------------
__global__ void k_attn(const unsigned short* __restrict__ Qb,
                       const unsigned short* __restrict__ Kb,
                       const unsigned short* __restrict__ Vt,
                       const int*   __restrict__ rel_idx,
                       const float* __restrict__ rpb,
                       const float* __restrict__ amask,
                       unsigned short* __restrict__ ao) {
    __shared__ unsigned short ldsP[8 * 1024];     // per-wave 16x64 bf16 P tile
    int wloc = threadIdx.x >> 5;
    int lane = threadIdx.x & 31;
    int id = blockIdx.x * 8 + wloc;               // < 65536
    int mt = id & 3, head = (id >> 2) & 7, w = id >> 5;
    int wl = w & 63;                              // window-in-image for mask
    int row = lane & 15, hi = lane >> 4;

    size_t hb = (size_t)(w * HEADS_ + head);
    const unsigned short* Qp = Qb + hb * 64 * HD_;
    const unsigned short* Kp = Kb + hb * 64 * HD_;
    const unsigned short* Vp = Vt + hb * HD_ * 64;

    // --- Q A-fragment (16 rows x K=32), single k-step since head_dim = 32 ---
    bf16x8 alo = *reinterpret_cast<const bf16x8*>(Qp + (mt * 16 + row) * HD_ + hi * 16);
    bf16x8 ahi = *reinterpret_cast<const bf16x8*>(Qp + (mt * 16 + row) * HD_ + 8 + hi * 16);
    bf16x16 a;
    #pragma unroll
    for (int i = 0; i < 8; i++) { a[i] = alo[i]; a[i + 8] = ahi[i]; }

    // --- S = Q K^T : B[k][n] = K[n][k], contiguous 32B per lane ---
    f32x8 st[4];
    #pragma unroll
    for (int t = 0; t < 4; t++) {
        bf16x16 bk = *reinterpret_cast<const bf16x16*>(Kp + (t * 16 + row) * HD_ + hi * 16);
        f32x8 z = {};
        st[t] = __builtin_amdgcn_wmma_f32_16x16x32_bf16(
            false, a, false, bk, (short)0, z, false, false);
    }

    // --- bias + shift-mask + key-pad mask, row softmax ---
    unsigned short* Pl = ldsP + wloc * 1024;
    #pragma unroll
    for (int r = 0; r < 8; r++) {
        int m = mt * 16 + r + 8 * hi;             // query token in window
        float mx = -3.0e38f;
        #pragma unroll
        for (int t = 0; t < 4; t++) {
            int n = t * 16 + row;                 // key token
            float v;
            if (m < N_ && n < N_)
                v = st[t][r] + rpb[rel_idx[m * N_ + n] * HEADS_ + head]
                             + amask[((size_t)wl * N_ + m) * N_ + n];
            else
                v = -1.0e9f;                      // padded keys -> 0 prob
            st[t][r] = v;
            mx = fmaxf(mx, v);
        }
        mx = half_red_max(mx);
        float sm = 0.f;
        #pragma unroll
        for (int t = 0; t < 4; t++) {
            float e = __expf(st[t][r] - mx);
            st[t][r] = e;
            sm += e;
        }
        sm = half_red_add(sm);
        float inv = 1.0f / sm;                    // sm > 0 always (exp > 0)
        #pragma unroll
        for (int t = 0; t < 4; t++) {
            int n = t * 16 + row;
            Pl[(r + 8 * hi) * 64 + n] = f2bf(st[t][r] * inv);
        }
    }

    asm volatile("s_wait_dscnt 0" ::: "memory");  // same-wave LDS RAW guard

    // --- O = P V : K=64 (two k-steps), N=32 (two n-tiles) ---
    // B[k][n] = V[k][n] = Vt[n][k] -> contiguous 32B per lane from Vt rows.
    f32x8 ot[2] = {};
    #pragma unroll
    for (int k0 = 0; k0 < 64; k0 += 32) {
        bf16x8 plo = *reinterpret_cast<const bf16x8*>(Pl + row * 64 + k0 + hi * 16);
        bf16x8 phi = *reinterpret_cast<const bf16x8*>(Pl + row * 64 + k0 + 8 + hi * 16);
        bf16x16 pa;
        #pragma unroll
        for (int i = 0; i < 8; i++) { pa[i] = plo[i]; pa[i + 8] = phi[i]; }
        #pragma unroll
        for (int nt = 0; nt < 2; nt++) {
            bf16x16 bv = *reinterpret_cast<const bf16x16*>(
                Vp + (nt * 16 + row) * 64 + k0 + hi * 16);
            ot[nt] = __builtin_amdgcn_wmma_f32_16x16x32_bf16(
                false, pa, false, bv, (short)0, ot[nt], false, false);
        }
    }

    // --- store O (windowed token-major, DIM cols), drop padded rows ---
    #pragma unroll
    for (int nt = 0; nt < 2; nt++)
        #pragma unroll
        for (int r = 0; r < 8; r++) {
            int m = mt * 16 + r + 8 * hi;
            if (m < N_) {
                size_t tok = (size_t)w * N_ + m;
                ao[tok * DIM_ + head * HD_ + nt * 16 + row] = f2bf(ot[nt][r]);
            }
        }
}

// ---------------------------------------------------------------------------
// K4: proj GEMM + window reverse + un-shift (via omap) + residual -> f32 d_out
// ---------------------------------------------------------------------------
__global__ void k_proj(const unsigned short* __restrict__ ao,
                       const unsigned short* __restrict__ Wp,
                       const float* __restrict__ bias,
                       const float* __restrict__ x0,
                       const int* __restrict__ omap,
                       float* __restrict__ xres) {
    int wave = blockIdx.x * 8 + (threadIdx.x >> 5);
    int lane = threadIdx.x & 31;
    int nt = wave & 3, mt = wave >> 2;            // 4 = 256/64
    int m0 = mt * 32, n0 = nt * 64;
    f32x8 acc[2][4] = {};
    gemm_tile<DIM_, 2>(ao + (size_t)m0 * DIM_, Wp + (size_t)n0 * DIM_, acc, lane);
    int row = lane & 15, hi = lane >> 4;
    int j0 = n0 + row;
    #pragma unroll
    for (int mi = 0; mi < 2; mi++)
        #pragma unroll
        for (int r = 0; r < 8; r++) {
            int m = m0 + mi * 16 + r + 8 * hi;
            size_t base = (size_t)omap[m] * DIM_; // scatter back to image order
            #pragma unroll
            for (int t = 0; t < 4; t++) {
                int j = j0 + t * 16;
                xres[base + j] = x0[base + j] + acc[mi][t][r] + bias[j];
            }
        }
}

// ---------------------------------------------------------------------------
// K6: fc1 GEMM + exact GELU -> bf16 hidden (100352 x 1024)
// ---------------------------------------------------------------------------
__global__ void k_fc1(const unsigned short* __restrict__ h2,
                      const unsigned short* __restrict__ W1,
                      const float* __restrict__ bias,
                      unsigned short* __restrict__ hid) {
    int wave = blockIdx.x * 8 + (threadIdx.x >> 5);
    int lane = threadIdx.x & 31;
    int nt = wave & 15, mt = wave >> 4;           // 16 = 1024/64
    int m0 = mt * 32, n0 = nt * 64;
    f32x8 acc[2][4] = {};
    gemm_tile<DIM_, 2>(h2 + (size_t)m0 * DIM_, W1 + (size_t)n0 * DIM_, acc, lane);
    int row = lane & 15, hi = lane >> 4;
    int j0 = n0 + row;
    #pragma unroll
    for (int mi = 0; mi < 2; mi++)
        #pragma unroll
        for (int r = 0; r < 8; r++) {
            int m = m0 + mi * 16 + r + 8 * hi;
            unsigned short* dst = hid + (size_t)m * HID_;
            #pragma unroll
            for (int t = 0; t < 4; t++) {
                int j = j0 + t * 16;
                float v = acc[mi][t][r] + bias[j];
                v = 0.5f * v * (1.0f + erff(v * 0.70710678118654752f));  // exact GELU
                dst[j] = f2bf(v);
            }
        }
}

// ---------------------------------------------------------------------------
// K7: fc2 GEMM (K=1024) + second residual accumulate into d_out
// ---------------------------------------------------------------------------
__global__ void k_fc2(const unsigned short* __restrict__ hid,
                      const unsigned short* __restrict__ W2,
                      const float* __restrict__ bias,
                      float* __restrict__ xres) {
    int wave = blockIdx.x * 8 + (threadIdx.x >> 5);
    int lane = threadIdx.x & 31;
    int nt = wave & 3, mt = wave >> 2;            // 4 = 256/64
    int m0 = mt * 32, n0 = nt * 64;
    f32x8 acc[2][4] = {};
    gemm_tile<HID_, 2>(hid + (size_t)m0 * HID_, W2 + (size_t)n0 * HID_, acc, lane);
    int row = lane & 15, hi = lane >> 4;
    int j0 = n0 + row;
    #pragma unroll
    for (int mi = 0; mi < 2; mi++)
        #pragma unroll
        for (int r = 0; r < 8; r++) {
            int m = m0 + mi * 16 + r + 8 * hi;
            float* dst = xres + (size_t)m * DIM_;
            #pragma unroll
            for (int t = 0; t < 4; t++) {
                int j = j0 + t * 16;
                dst[j] += acc[mi][t][r] + bias[j];
            }
        }
}

// ---------------------------------------------------------------------------
// Host launcher
// ---------------------------------------------------------------------------
extern "C" void kernel_launch(void* const* d_in, const int* in_sizes, int n_in,
                              void* d_out, int out_size, void* d_ws, size_t ws_size,
                              hipStream_t stream) {
    (void)in_sizes; (void)n_in; (void)out_size; (void)ws_size;

    const float* x      = (const float*)d_in[0];
    const float* n1w    = (const float*)d_in[1];
    const float* n1b    = (const float*)d_in[2];
    const float* qkvw_f = (const float*)d_in[3];
    const float* qkvb   = (const float*)d_in[4];
    const float* rpb    = (const float*)d_in[5];
    const float* projw_f= (const float*)d_in[6];
    const float* projb  = (const float*)d_in[7];
    const float* n2w    = (const float*)d_in[8];
    const float* n2b    = (const float*)d_in[9];
    const float* fc1w_f = (const float*)d_in[10];
    const float* fc1b   = (const float*)d_in[11];
    const float* fc2w_f = (const float*)d_in[12];
    const float* fc2b   = (const float*)d_in[13];
    const int*   relidx = (const int*)d_in[14];
    const float* amask  = (const float*)d_in[15];
    float* out = (float*)d_out;

    // workspace partition (256B aligned)
    char* ws = (char*)d_ws;
    size_t off = 0;
    auto take = [&](size_t bytes) -> char* {
        char* p = ws + off;
        off = (off + bytes + 255) & ~(size_t)255;
        return p;
    };
    unsigned short* qkvw  = (unsigned short*)take((size_t)768 * 256 * 2);
    unsigned short* projw = (unsigned short*)take((size_t)256 * 256 * 2);
    unsigned short* fc1w  = (unsigned short*)take((size_t)1024 * 256 * 2);
    unsigned short* fc2w  = (unsigned short*)take((size_t)256 * 1024 * 2);
    int*            wn    = (int*)take((size_t)TOK_ * 4);
    int*            omap  = (int*)take((size_t)TOK_ * 4);
    unsigned short* xwin  = (unsigned short*)take((size_t)TOK_ * DIM_ * 2);
    unsigned short* Qb    = (unsigned short*)take((size_t)WT_ * HEADS_ * 64 * HD_ * 2);
    unsigned short* Kb    = (unsigned short*)take((size_t)WT_ * HEADS_ * 64 * HD_ * 2);
    unsigned short* Vt    = (unsigned short*)take((size_t)WT_ * HEADS_ * HD_ * 64 * 2);
    unsigned short* ao    = (unsigned short*)take((size_t)TOK_ * DIM_ * 2);
    unsigned short* h2    = (unsigned short*)take((size_t)TOK_ * DIM_ * 2);
    unsigned short* hid   = (unsigned short*)take((size_t)TOK_ * HID_ * 2);

    dim3 blk(256);

    // index maps + weight conversions
    k_maps<<<dim3((TOK_ + 255) / 256), blk, 0, stream>>>(wn, omap);
    k_cvt<<<dim3((768 * 256 + 255) / 256), blk, 0, stream>>>(qkvw_f, qkvw, 768 * 256);
    k_cvt<<<dim3((256 * 256 + 255) / 256), blk, 0, stream>>>(projw_f, projw, 256 * 256);
    k_cvt<<<dim3((1024 * 256 + 255) / 256), blk, 0, stream>>>(fc1w_f, fc1w, 1024 * 256);
    k_cvt<<<dim3((256 * 1024 + 255) / 256), blk, 0, stream>>>(fc2w_f, fc2w, 256 * 1024);

    // LN1 + shift + window partition (gather via omap), bf16
    k_ln<<<dim3(TOK_ / 8), blk, 0, stream>>>(x, n1w, n1b, xwin, omap, 1);

    // QKV projection -> padded Q/K + transposed V
    k_qkv<<<dim3((3136 * 12) / 8), blk, 0, stream>>>(xwin, qkvw, qkvb, wn, Qb, Kb, Vt);

    // windowed attention
    k_attn<<<dim3(65536 / 8), blk, 0, stream>>>(Qb, Kb, Vt, relidx, rpb, amask, ao);

    // proj + window reverse + un-shift + residual -> d_out (f32)
    k_proj<<<dim3((3136 * 4) / 8), blk, 0, stream>>>(ao, projw, projb, x, omap, out);

    // LN2 (image order), bf16
    k_ln<<<dim3(TOK_ / 8), blk, 0, stream>>>(out, n2w, n2b, h2, omap, 0);

    // MLP
    k_fc1<<<dim3((3136 * 16) / 8), blk, 0, stream>>>(h2, fc1w, fc1b, hid);
    k_fc2<<<dim3((3136 * 4) / 8), blk, 0, stream>>>(hid, fc2w, fc2b, out);
}
